// GroupedQueryAttention_24309514895717
// MI455X (gfx1250) — matrix operand order repaired and checked by reading, not compiled, not run on hardware
//
#include <hip/hip_runtime.h>

typedef __attribute__((ext_vector_type(16))) __bf16 v16bf;
typedef __attribute__((ext_vector_type(8)))  float  v8f;
typedef __attribute__((ext_vector_type(4)))  unsigned int u32x4;
typedef __attribute__((ext_vector_type(8)))  int i32x8;
typedef __attribute__((ext_vector_type(4)))  int i32x4;

constexpr int Bc = 2, Sc = 2048, Dc = 1024, Hc = 16, KVHc = 4, HDc = 64;
constexpr int KVD = KVHc * HDc;      // 256
constexpr float SCALEc = 0.125f;     // 64^-0.5

// ---- WMMA fragment K-striping per CDNA5 ISA tables (16-bit elements) ----
__device__ __forceinline__ int frag_k0_a(int v, int hi) {
  return (v < 4 ? 2 * v : 8 + 2 * v) + (hi ? 8 : 0);
}
__device__ __forceinline__ int frag_k0_b(int v, int hi) {
  return (v < 4 ? 2 * v : 8 + 2 * v) + (hi ? 8 : 0);
}

// ---- CDNA5 async global->LDS (ISA 15.18.3 op 98), tracked by ASYNCcnt ----
__device__ __forceinline__ void async_load_b128(unsigned lds_off, const void* g) {
  asm volatile("global_load_async_to_lds_b128 %0, %1, off"
               :: "v"(lds_off), "v"((unsigned long long)(uintptr_t)g)
               : "memory");
}
__device__ __forceinline__ void wait_async0() {
  asm volatile("s_wait_asynccnt 0x0" ::: "memory");
}
__device__ __forceinline__ void wait_async3() {
  asm volatile("s_wait_asynccnt 0x3" ::: "memory");
}
__device__ __forceinline__ unsigned lds_off(const void* p) {
  return (unsigned)(uintptr_t)p;  // generic LDS addr: low 32 bits = LDS offset
}

// ---- CDNA5 Tensor Data Mover: 2D bf16 tile -> LDS (ISA ch.8 D# layout) ----
#if __has_builtin(__builtin_amdgcn_tensor_load_to_lds)
#define HAVE_TDM 1
#else
#define HAVE_TDM 0
#endif

#if HAVE_TDM
// Loads tile_h rows x tile_w bf16 from row-stride `stride_elems`, padding the
// LDS destination by 4 DWORDs after every 16 DWORDs stored (row = 32 bf16 =
// 16 DW), i.e. LDS row stride = 40 bf16 == ATS/BTS.
__device__ __forceinline__ void tdm_load_2d(unsigned lds, const void* g,
                                            int tile_w, int tile_h,
                                            int stride_elems) {
  u32x4 g0 = {0u, 0u, 0u, 0u};
  i32x8 g1 = {0, 0, 0, 0, 0, 0, 0, 0};
  i32x4 gz = {0, 0, 0, 0};
  unsigned long long ga = (unsigned long long)(uintptr_t)g;
  g0[0] = 1u;                                   // count=1 (valid user D#)
  g0[1] = lds;                                  // lds_addr (bytes)
  g0[2] = (unsigned)ga;                         // global_addr[31:0]
  g0[3] = (unsigned)((ga >> 32) & 0x01FFFFFFu)  // global_addr[56:32]
          | (2u << 30);                         // type = 2 ("image")
  g1[0] = (1 << 16)                             // data_size = 1 -> 2 bytes
          | (1 << 20)                           // pad_enable
          | (3 << 22)                           // pad_interval code 3 -> 16 DW
          | (3 << 25);                          // pad_amount  code 3 -> 4 DW
  g1[1] = (stride_elems & 0xFFFF) << 16;        // tensor_dim0[15:0]
  g1[2] = (stride_elems >> 16) & 0xFFFF;        // tensor_dim0[31:16] | dim1 lo
  g1[3] = 0x4000 | ((tile_w & 0xFFFF) << 16);   // tensor_dim1 hi (1<<30) | tile_dim0
  g1[4] = tile_h & 0xFFFF;                      // tile_dim1 (tile_dim2 = 0)
  g1[5] = stride_elems;                         // tensor_dim0_stride[31:0]
  g1[6] = 0;
  g1[7] = 0;
#if __clang_major__ >= 23
  i32x8 g4 = {0, 0, 0, 0, 0, 0, 0, 0};
  __builtin_amdgcn_tensor_load_to_lds(g0, g1, gz, gz, g4, 0);
#else
  __builtin_amdgcn_tensor_load_to_lds(g0, g1, gz, gz, 0);
#endif
}
#endif

// ---- DPP-based 16-lane reductions (no LDS traffic) ----
#if __has_builtin(__builtin_amdgcn_update_dpp)
template <int CTRL>
__device__ __forceinline__ float dpp_movf(float x) {
  return __int_as_float(
      __builtin_amdgcn_update_dpp(0, __float_as_int(x), CTRL, 0xf, 0xf, true));
}
__device__ __forceinline__ float red16_max(float x) {
  x = fmaxf(x, dpp_movf<0xB1>(x));   // quad_perm(1,0,3,2)  : xor1
  x = fmaxf(x, dpp_movf<0x4E>(x));   // quad_perm(2,3,0,1)  : xor2
  x = fmaxf(x, dpp_movf<0x141>(x));  // row_half_mirror     : ~xor4
  x = fmaxf(x, dpp_movf<0x140>(x));  // row_mirror          : ~xor8
  return x;
}
__device__ __forceinline__ float red16_sum(float x) {
  x += dpp_movf<0xB1>(x);
  x += dpp_movf<0x4E>(x);
  x += dpp_movf<0x141>(x);
  x += dpp_movf<0x140>(x);
  return x;
}
#else
__device__ __forceinline__ float red16_max(float x) {
  for (int off = 1; off < 16; off <<= 1) x = fmaxf(x, __shfl_xor(x, off, 32));
  return x;
}
__device__ __forceinline__ float red16_sum(float x) {
  for (int off = 1; off < 16; off <<= 1) x += __shfl_xor(x, off, 32);
  return x;
}
#endif

// ---------------------------------------------------------------- convert --
__global__ __launch_bounds__(256) void cvt_f32_bf16(const float* __restrict__ in,
                                                    __bf16* __restrict__ out, int n) {
  int i = (blockIdx.x * 256 + threadIdx.x) * 4;
  if (i + 3 < n) {
    float4 v = *(const float4*)(in + i);
    union { __bf16 e[4]; unsigned long long u; } t;
    t.e[0] = (__bf16)v.x; t.e[1] = (__bf16)v.y;
    t.e[2] = (__bf16)v.z; t.e[3] = (__bf16)v.w;
    *(unsigned long long*)(out + i) = t.u;
  }
}

// --------------------------------------------------------- GEMM C = A * B^T --
// Block 256 thr / 8 waves; block tile 128(M) x 64(N); wave macro-tile 32x32
// (2x2 WMMA accs). K staged 32 wide, double-buffered: staged by the Tensor
// Data Mover (one D# per tile, wave 0 issues) or async-to-LDS fallback.
constexpr int ATS = 40, BTS = 40;  // padded LDS strides (bf16) == TDM pad result

template <typename OutT>
__global__ __launch_bounds__(256) void gemm_bf16_wmma(const __bf16* __restrict__ A,
                                                      const __bf16* __restrict__ Bw,
                                                      OutT* __restrict__ C,
                                                      int M, int N, int K) {
  __shared__ __bf16 As[2][128 * ATS];
  __shared__ __bf16 Bs[2][64 * BTS];
  const int tid = threadIdx.x;
  const int lane = tid & 31;
  const int w = tid >> 5;
  const int hi = lane >> 4;
  const int mn = lane & 15;
  const int rowBase = blockIdx.y * 128;
  const int colBase = blockIdx.x * 64;
  const int wr = (w & 3) * 32;
  const int wc = (w >> 2) * 32;

  auto issue = [&](int buf, int k0) {
#if HAVE_TDM
    if (w == 0) {  // TDM ignores EXEC; descriptors are wave-uniform
      tdm_load_2d(lds_off(&As[buf][0]), A + (size_t)rowBase * K + k0, 32, 128, K);
      tdm_load_2d(lds_off(&Bs[buf][0]), Bw + (size_t)colBase * K + k0, 32, 64, K);
    }
#else
#pragma unroll
    for (int i = 0; i < 2; ++i) {  // A tile: 128x32 = 512 b128 chunks
      int id = tid + i * 256;
      int r = id >> 2, c = (id & 3) * 8;
      async_load_b128(lds_off(&As[buf][r * ATS + c]),
                      A + (size_t)(rowBase + r) * K + k0 + c);
    }
    {  // B tile: 64x32 = 256 b128 chunks
      int r = tid >> 2, c = (tid & 3) * 8;
      async_load_b128(lds_off(&Bs[buf][r * BTS + c]),
                      Bw + (size_t)(colBase + r) * K + k0 + c);
    }
#endif
  };

  v8f a00 = {}, a01 = {}, a10 = {}, a11 = {};
  const int nk = K >> 5;
  issue(0, 0);
  for (int i = 0; i < nk; ++i) {
    __syncthreads();  // readers of the buffer being refilled are done
    if (i + 1 < nk) {
      issue((i + 1) & 1, (i + 1) << 5);
#if HAVE_TDM
      __builtin_amdgcn_s_wait_tensorcnt(2);  // in-order: current stage done
#else
      wait_async3();
#endif
    } else {
#if HAVE_TDM
      __builtin_amdgcn_s_wait_tensorcnt(0);
#else
      wait_async0();
#endif
    }
    __syncthreads();  // current stage visible to all waves

    const __bf16* Ab = As[i & 1];
    const __bf16* Bb = Bs[i & 1];
    union { v16bf v; unsigned u[8]; } fa0, fa1, fb0, fb1;
#pragma unroll
    for (int v = 0; v < 8; ++v) {
      const int ka = frag_k0_a(v, hi), kb = frag_k0_b(v, hi);
      fa0.u[v] = *(const unsigned*)(&Ab[(wr + mn) * ATS + ka]);
      fa1.u[v] = *(const unsigned*)(&Ab[(wr + 16 + mn) * ATS + ka]);
      fb0.u[v] = *(const unsigned*)(&Bb[(wc + mn) * BTS + kb]);
      fb1.u[v] = *(const unsigned*)(&Bb[(wc + 16 + mn) * BTS + kb]);
    }
    a00 = __builtin_amdgcn_wmma_f32_16x16x32_bf16(false, fa0.v, false, fb0.v, (short)0, a00, false, false);
    a01 = __builtin_amdgcn_wmma_f32_16x16x32_bf16(false, fa0.v, false, fb1.v, (short)0, a01, false, false);
    a10 = __builtin_amdgcn_wmma_f32_16x16x32_bf16(false, fa1.v, false, fb0.v, (short)0, a10, false, false);
    a11 = __builtin_amdgcn_wmma_f32_16x16x32_bf16(false, fa1.v, false, fb1.v, (short)0, a11, false, false);
  }

#pragma unroll
  for (int r = 0; r < 8; ++r) {  // C-layout: M = r + 8*hi, N = lane&15
    const int r0 = rowBase + wr + 8 * hi + r;
    const int r1 = r0 + 16;
    const int c0 = colBase + wc + mn;
    const int c1 = c0 + 16;
    C[(size_t)r0 * N + c0] = (OutT)a00[r];
    C[(size_t)r0 * N + c1] = (OutT)a01[r];
    C[(size_t)r1 * N + c0] = (OutT)a10[r];
    C[(size_t)r1 * N + c1] = (OutT)a11[r];
  }
}

// --------------------------------------------------- flash GQA (bf16 WMMA) --
constexpr int KTS = 40, VTS = 40, PTS = 40;

__global__ __launch_bounds__(256) void gqa_flash_wmma(const __bf16* __restrict__ Q,
                                                      const __bf16* __restrict__ Kb,
                                                      const __bf16* __restrict__ Vb,
                                                      const int* __restrict__ maskp,
                                                      __bf16* __restrict__ Ao) {
  __shared__ __bf16 Ks[32 * KTS];      // [kv][hd]
  __shared__ __bf16 Vst[64 * VTS];     // [hd][kv] (transposed for B-operand)
  __shared__ __bf16 Pw[8 * 16 * PTS];  // per-wave P scratch

  const int tid = threadIdx.x;
  const int lane = tid & 31;
  const int w = tid >> 5;
  const int hi = lane >> 4;
  const int mn = lane & 15;
  const int b = blockIdx.z;
  const int h = blockIdx.y;
  const int kvh = h >> 2;
  const int qbase = blockIdx.x * 128;
  const int q = qbase + w * 16 + mn;

  union U16 { v16bf v; unsigned u[8]; };
  U16 fq0, fq1;
  {
    const __bf16* qp = Q + ((size_t)(b * Sc + q)) * Dc + h * HDc;
#pragma unroll
    for (int v = 0; v < 8; ++v) {
      int kk = frag_k0_a(v, hi);
      fq0.u[v] = *(const unsigned*)(qp + kk);
      fq1.u[v] = *(const unsigned*)(qp + 32 + kk);
    }
  }

  float mrow[8], lrow[8];
  v8f o0 = {}, o1 = {}, o2 = {}, o3 = {};
#pragma unroll
  for (int r = 0; r < 8; ++r) { mrow[r] = -__builtin_inff(); lrow[r] = 0.f; }

  __bf16* Pb = &Pw[w * 16 * PTS];
  const int kr = tid >> 3, kc = (tid & 7) * 8;     // K-tile chunk (async)
  const int vkv = tid & 31, vhd = (tid >> 5) * 8;  // V-tile transpose lanes

  for (int kv0 = 0; kv0 < Sc; kv0 += 32) {
    __syncthreads();  // previous iteration's frag reads done
    // K tile 32x64 via async DMA to LDS
    async_load_b128(lds_off(&Ks[kr * KTS + kc]),
                    Kb + ((size_t)(b * Sc + kv0 + kr)) * KVD + kvh * HDc + kc);
    // V tile, transposed into Vst[hd][kv]
    {
      union { uint4 u4; __bf16 e[8]; } t;
      t.u4 = *(const uint4*)(Vb + ((size_t)(b * Sc + kv0 + vkv)) * KVD + kvh * HDc + vhd);
#pragma unroll
      for (int i = 0; i < 8; ++i) Vst[(vhd + i) * VTS + vkv] = t.e[i];
    }
    wait_async0();
    __syncthreads();

    // ---- scores: two 16(q) x 16(kv) C-tiles, K-dim = HD = 64
    v8f s0 = {}, s1 = {};
    {
      U16 fk;
#pragma unroll
      for (int v = 0; v < 8; ++v)
        fk.u[v] = *(const unsigned*)(&Ks[mn * KTS + frag_k0_b(v, hi)]);
      s0 = __builtin_amdgcn_wmma_f32_16x16x32_bf16(false, fq0.v, false, fk.v, (short)0, s0, false, false);
#pragma unroll
      for (int v = 0; v < 8; ++v)
        fk.u[v] = *(const unsigned*)(&Ks[mn * KTS + 32 + frag_k0_b(v, hi)]);
      s0 = __builtin_amdgcn_wmma_f32_16x16x32_bf16(false, fq1.v, false, fk.v, (short)0, s0, false, false);
#pragma unroll
      for (int v = 0; v < 8; ++v)
        fk.u[v] = *(const unsigned*)(&Ks[(16 + mn) * KTS + frag_k0_b(v, hi)]);
      s1 = __builtin_amdgcn_wmma_f32_16x16x32_bf16(false, fq0.v, false, fk.v, (short)0, s1, false, false);
#pragma unroll
      for (int v = 0; v < 8; ++v)
        fk.u[v] = *(const unsigned*)(&Ks[(16 + mn) * KTS + 32 + frag_k0_b(v, hi)]);
      s1 = __builtin_amdgcn_wmma_f32_16x16x32_bf16(false, fq1.v, false, fk.v, (short)0, s1, false, false);
    }

    // ---- scale + mask
    const int mk0 = maskp[b * Sc + kv0 + mn];
    const int mk1 = maskp[b * Sc + kv0 + 16 + mn];
#pragma unroll
    for (int r = 0; r < 8; ++r) {
      s0[r] = mk0 ? s0[r] * SCALEc : -__builtin_inff();
      s1[r] = mk1 ? s1[r] * SCALEc : -__builtin_inff();
    }

    // ---- online softmax (DPP row reductions)
    float p0[8], p1[8], rs[8];
#pragma unroll
    for (int r = 0; r < 8; ++r) {
      float rm = red16_max(fmaxf(s0[r], s1[r]));
      float mnew = fmaxf(mrow[r], rm);
      float alpha = (mrow[r] > -__builtin_inff()) ? __expf(mrow[r] - mnew) : 0.f;
      float e0 = (mnew > -__builtin_inff()) ? __expf(s0[r] - mnew) : 0.f;
      float e1 = (mnew > -__builtin_inff()) ? __expf(s1[r] - mnew) : 0.f;
      p0[r] = e0; p1[r] = e1;
      rs[r] = e0 + e1;
      mrow[r] = mnew;
      lrow[r] *= alpha;
      o0[r] *= alpha; o1[r] *= alpha; o2[r] *= alpha; o3[r] *= alpha;
    }
#pragma unroll
    for (int r = 0; r < 8; ++r) lrow[r] += red16_sum(rs[r]);

    // ---- P: C-layout -> per-wave LDS -> A-operand layout
#pragma unroll
    for (int r = 0; r < 8; ++r) {
      Pb[(r + 8 * hi) * PTS + mn] = (__bf16)p0[r];
      Pb[(r + 8 * hi) * PTS + 16 + mn] = (__bf16)p1[r];
    }
    asm volatile("s_wait_dscnt 0x0" ::: "memory");
    U16 fp;
#pragma unroll
    for (int v = 0; v < 8; ++v)
      fp.u[v] = *(const unsigned*)(&Pb[mn * PTS + frag_k0_a(v, hi)]);

    // ---- PV: O[16x64] += P[16x32] x V[32x64]
    U16 fv;
#pragma unroll
    for (int v = 0; v < 8; ++v)
      fv.u[v] = *(const unsigned*)(&Vst[(0 + mn) * VTS + frag_k0_b(v, hi)]);
    o0 = __builtin_amdgcn_wmma_f32_16x16x32_bf16(false, fp.v, false, fv.v, (short)0, o0, false, false);
#pragma unroll
    for (int v = 0; v < 8; ++v)
      fv.u[v] = *(const unsigned*)(&Vst[(16 + mn) * VTS + frag_k0_b(v, hi)]);
    o1 = __builtin_amdgcn_wmma_f32_16x16x32_bf16(false, fp.v, false, fv.v, (short)0, o1, false, false);
#pragma unroll
    for (int v = 0; v < 8; ++v)
      fv.u[v] = *(const unsigned*)(&Vst[(32 + mn) * VTS + frag_k0_b(v, hi)]);
    o2 = __builtin_amdgcn_wmma_f32_16x16x32_bf16(false, fp.v, false, fv.v, (short)0, o2, false, false);
#pragma unroll
    for (int v = 0; v < 8; ++v)
      fv.u[v] = *(const unsigned*)(&Vst[(48 + mn) * VTS + frag_k0_b(v, hi)]);
    o3 = __builtin_amdgcn_wmma_f32_16x16x32_bf16(false, fp.v, false, fv.v, (short)0, o3, false, false);
  }

  // ---- epilogue
#pragma unroll
  for (int r = 0; r < 8; ++r) {
    float rl = lrow[r] > 0.f ? 1.f / lrow[r] : 0.f;
    int gq = qbase + w * 16 + 8 * hi + r;
    size_t base = ((size_t)(b * Sc + gq)) * Dc + h * HDc;
    Ao[base + mn]      = (__bf16)(o0[r] * rl);
    Ao[base + 16 + mn] = (__bf16)(o1[r] * rl);
    Ao[base + 32 + mn] = (__bf16)(o2[r] * rl);
    Ao[base + 48 + mn] = (__bf16)(o3[r] * rl);
  }
}

// ------------------------------------------------------------------ launch --
extern "C" void kernel_launch(void* const* d_in, const int* in_sizes, int n_in,
                              void* d_out, int out_size, void* d_ws, size_t ws_size,
                              hipStream_t stream) {
  const float* x  = (const float*)d_in[0];
  const int*   mk = (const int*)d_in[1];
  const float* Wq = (const float*)d_in[2];
  const float* Wk = (const float*)d_in[3];
  const float* Wv = (const float*)d_in[4];
  const float* Wo = (const float*)d_in[5];
  float* out = (float*)d_out;

  char* ws = (char*)d_ws;
  size_t off = 0;
  auto take = [&](size_t elems) {
    __bf16* p = (__bf16*)(ws + off);
    off += ((elems * 2 + 255) & ~(size_t)255);
    return p;
  };
  const int M = Bc * Sc;  // 4096
  __bf16* Xbf = take((size_t)M * Dc);
  __bf16* Wqb = take((size_t)Dc * Dc);
  __bf16* Wkb = take((size_t)KVD * Dc);
  __bf16* Wvb = take((size_t)KVD * Dc);
  __bf16* Wob = take((size_t)Dc * Dc);
  __bf16* Qb  = take((size_t)M * Dc);
  __bf16* Kb  = take((size_t)M * KVD);
  __bf16* Vb  = take((size_t)M * KVD);
  __bf16* Ab  = take((size_t)M * Dc);

  auto cvt = [&](const float* in, __bf16* o, int n) {
    cvt_f32_bf16<<<dim3((n / 4 + 255) / 256), 256, 0, stream>>>(in, o, n);
  };
  cvt(x,  Xbf, M * Dc);
  cvt(Wq, Wqb, Dc * Dc);
  cvt(Wk, Wkb, KVD * Dc);
  cvt(Wv, Wvb, KVD * Dc);
  cvt(Wo, Wob, Dc * Dc);

  gemm_bf16_wmma<__bf16><<<dim3(Dc / 64, M / 128), 256, 0, stream>>>(Xbf, Wqb, Qb, M, Dc, Dc);
  gemm_bf16_wmma<__bf16><<<dim3(KVD / 64, M / 128), 256, 0, stream>>>(Xbf, Wkb, Kb, M, KVD, Dc);
  gemm_bf16_wmma<__bf16><<<dim3(KVD / 64, M / 128), 256, 0, stream>>>(Xbf, Wvb, Vb, M, KVD, Dc);

  gqa_flash_wmma<<<dim3(Sc / 128, Hc, Bc), 256, 0, stream>>>(Qb, Kb, Vb, mk, Ab);

  gemm_bf16_wmma<float><<<dim3(Dc / 64, M / 128), 256, 0, stream>>>(Ab, Wob, out, M, Dc, Dc);
}
